// SlidingWindowMultiHeadAttention_11879879540963
// MI455X (gfx1250) — matrix operand order repaired and checked by reading, not compiled, not run on hardware
//
#include <hip/hip_runtime.h>
#include <hip/hip_bf16.h>

typedef __attribute__((ext_vector_type(16))) __bf16 v16bf;
typedef __attribute__((ext_vector_type(8)))  __bf16 v8bf;
typedef __attribute__((ext_vector_type(8)))  float  v8f;
typedef __attribute__((ext_vector_type(4))) unsigned int u32x4;
typedef __attribute__((ext_vector_type(8))) int          i32x8;
typedef __attribute__((ext_vector_type(4))) int          i32x4;

#define DIM   1024
#define SEQ   2048
#define NB    2
#define NH    16
#define HD    64
#define WIN   256
#define KC    64          // K-chunk staged in LDS
#define NEG_BIG (-3.0e38f)

#if defined(__has_builtin)
#if __has_builtin(__builtin_amdgcn_tensor_load_to_lds) && __has_builtin(__builtin_amdgcn_s_wait_tensorcnt)
#define USE_TDM 1
#endif
#endif

__device__ __forceinline__ unsigned short f2bf(float f) {
  unsigned int u = __builtin_bit_cast(unsigned int, f);
  unsigned int r = u + 0x7FFFu + ((u >> 16) & 1u);   // round-to-nearest-even
  return (unsigned short)(r >> 16);
}

__device__ __forceinline__ v8f wmma_bf16(v16bf a, v16bf b, v8f c) {
  return __builtin_amdgcn_wmma_f32_16x16x32_bf16(false, a, false, b,
                                                 (short)0, c, false, false);
}

#if USE_TDM
// 2-D TDM load: tile_d1 rows x tile_d0 elements (2-byte), row stride = stride0
// elements, from `gaddr` (tile start) into LDS offset lds_off (contiguous).
// D# layout per CDNA5 ISA 8.3/8.4.  This toolchain's builtin takes 6 args
// (g0 v4u, g1 v8i, g2 v4i, g3 v4i, g4 v8i, cpol).
__device__ __forceinline__ void tdm_load_2d(unsigned lds_off, const void* gaddr,
                                            unsigned tensor_d0, unsigned tensor_d1,
                                            unsigned stride0,
                                            unsigned tile_d0, unsigned tile_d1) {
  unsigned long long ga = (unsigned long long)(size_t)gaddr;
  u32x4 g0;
  g0[0] = 1u;                                            // count=1 (valid user desc)
  g0[1] = lds_off;                                       // lds_addr
  g0[2] = (unsigned)(ga & 0xFFFFFFFFu);                  // global_addr[31:0]
  g0[3] = (unsigned)((ga >> 32) & 0x1FFFFFFu) | (2u << 30); // addr[56:32] | type=2
  i32x8 g1;
  g1[0] = (int)(1u << 16);                               // data_size=1 -> 2 bytes
  g1[1] = (int)((tensor_d0 & 0xFFFFu) << 16);            // tensor_dim0[15:0]
  g1[2] = (int)(((tensor_d0 >> 16) & 0xFFFFu) | ((tensor_d1 & 0xFFFFu) << 16));
  g1[3] = (int)(((tensor_d1 >> 16) & 0xFFFFu) | ((tile_d0 & 0xFFFFu) << 16));
  g1[4] = (int)(tile_d1 & 0xFFFFu);                      // tile_dim1 | tile_dim2=0
  g1[5] = (int)stride0;                                  // tensor_dim0_stride[31:0]
  g1[6] = 0;                                             // stride0 hi | stride1 lo
  g1[7] = 0;
  i32x4 z4 = {0, 0, 0, 0};
  i32x8 z8 = {0, 0, 0, 0, 0, 0, 0, 0};
  __builtin_amdgcn_tensor_load_to_lds(g0, g1, z4, z4, z8, 0);
}
#endif

// cooperative fallback copy: 128 rows x 64 bf16 elements into LDS (row-major)
__device__ __forceinline__ void coop_copy_tile(unsigned short* dst,
                                               const unsigned short* src,
                                               size_t srcStride) {
  int t = threadIdx.x;
  int row  = t & 127;
  int half = t >> 7;                         // 0..1, 32 elems (64B) each
  const unsigned short* s = src + (size_t)row * srcStride + half * 32;
  unsigned short* d = dst + row * KC + half * 32;
  *(v16bf*)(d)      = *(const v16bf*)(s);
  *(v16bf*)(d + 16) = *(const v16bf*)(s + 16);
}

// ---------------- conversion kernels ----------------
__global__ void k_cvt_x(const float* __restrict__ x, unsigned short* __restrict__ xb) {
  int i = blockIdx.x * blockDim.x + threadIdx.x;
  xb[i] = f2bf(x[i]);
}

// WT[n][k] = W[k][n]  (bf16), LDS-tiled so both sides are coalesced
__global__ __launch_bounds__(256) void k_cvt_wT(const float* __restrict__ w,
                                                unsigned short* __restrict__ wt) {
  __shared__ unsigned short t[32][33];
  int n0 = blockIdx.x * 32;
  int k0 = blockIdx.y * 32;
  int tx = threadIdx.x & 31;
  int ty = threadIdx.x >> 5;                 // 0..7
#pragma unroll
  for (int i = 0; i < 32; i += 8)
    t[ty + i][tx] = f2bf(w[(size_t)(k0 + ty + i) * DIM + n0 + tx]);
  __syncthreads();
#pragma unroll
  for (int i = 0; i < 32; i += 8)
    wt[(size_t)(n0 + ty + i) * DIM + k0 + tx] = t[tx][ty + i];
}

// ---------------- blocked GEMM body (shared by proj / outproj) ----------------
// Block computes 128x128 tile; 8 waves in 4(M)x2(N); each wave 32x64 (2x4 WMMA tiles).
__device__ __forceinline__ void gemm_128x128(
    const unsigned short* __restrict__ aSrc, size_t aStride, unsigned aRows,
    const unsigned short* __restrict__ bSrc, size_t bStride, unsigned bRows,
    int m0, int n0, unsigned short (*ldsA)[128 * KC], unsigned short (*ldsB)[128 * KC],
    v8f (&c)[2][4]) {
  int tid  = threadIdx.x;
  int wid  = tid >> 5;
  int lane = tid & 31;
  int waveM = wid >> 1;                      // 0..3
  int waveN = wid & 1;                       // 0..1
  int row  = lane & 15;
  int half = lane >> 4;
  int col  = lane & 15;
  int ka   = half * 16;
  int kbb  = half * 16;

  const int NKC = DIM / KC;                  // 16 chunks

  // stage chunk 0 into buffer 0
#if USE_TDM
  if (wid == 0) {
    tdm_load_2d((unsigned)(size_t)(void*)&ldsA[0][0], aSrc + (size_t)m0 * aStride,
                DIM, aRows, (unsigned)aStride, KC, 128);
    tdm_load_2d((unsigned)(size_t)(void*)&ldsB[0][0], bSrc + (size_t)n0 * bStride,
                DIM, bRows, (unsigned)bStride, KC, 128);
    __builtin_amdgcn_s_wait_tensorcnt(0);
  }
#else
  coop_copy_tile(&ldsA[0][0], aSrc + (size_t)m0 * aStride, aStride);
  coop_copy_tile(&ldsB[0][0], bSrc + (size_t)n0 * bStride, bStride);
#endif
  __syncthreads();

  int buf = 0;
  for (int ci = 0; ci < NKC; ++ci) {
    int kcNext = (ci + 1) * KC;
    if (ci + 1 < NKC) {
#if USE_TDM
      if (wid == 0) {
        tdm_load_2d((unsigned)(size_t)(void*)&ldsA[buf ^ 1][0],
                    aSrc + (size_t)m0 * aStride + kcNext, DIM, aRows,
                    (unsigned)aStride, KC, 128);
        tdm_load_2d((unsigned)(size_t)(void*)&ldsB[buf ^ 1][0],
                    bSrc + (size_t)n0 * bStride + kcNext, DIM, bRows,
                    (unsigned)bStride, KC, 128);
      }
#else
      coop_copy_tile(&ldsA[buf ^ 1][0], aSrc + (size_t)m0 * aStride + kcNext, aStride);
      coop_copy_tile(&ldsB[buf ^ 1][0], bSrc + (size_t)n0 * bStride + kcNext, bStride);
#endif
    }
    // compute 2 K-steps of 32 from current buffer
#pragma unroll
    for (int kk = 0; kk < KC; kk += 32) {
      v16bf a[2];
#pragma unroll
      for (int mt = 0; mt < 2; ++mt) {
        const unsigned short* ar =
            &ldsA[buf][(waveM * 32 + mt * 16 + row) * KC + kk];
        v8bf lo = *(const v8bf*)(ar + ka);
        v8bf hi = *(const v8bf*)(ar + ka + 8);
#pragma unroll
        for (int i = 0; i < 8; ++i) { a[mt][i] = lo[i]; a[mt][8 + i] = hi[i]; }
      }
#pragma unroll
      for (int t = 0; t < 4; ++t) {
        v16bf b = *(const v16bf*)&ldsB[buf][(waveN * 64 + t * 16 + col) * KC + kk + kbb];
        c[0][t] = wmma_bf16(a[0], b, c[0][t]);
        c[1][t] = wmma_bf16(a[1], b, c[1][t]);
      }
    }
#if USE_TDM
    if (wid == 0 && ci + 1 < NKC) __builtin_amdgcn_s_wait_tensorcnt(0);
#endif
    __syncthreads();
    buf ^= 1;
  }
}

// ---------------- fused QKV projection ----------------
// grid: 3 (which) x 32 (Mblk) x 8 (Nblk) = 768 blocks of 256 threads
__global__ __launch_bounds__(256) void k_proj(
    const unsigned short* __restrict__ xb,
    const unsigned short* __restrict__ wqt,
    const unsigned short* __restrict__ wkt,
    const unsigned short* __restrict__ wvt,
    unsigned short* __restrict__ qb,   // [b,h,s,64]
    unsigned short* __restrict__ kb,   // [b,h,s,64]
    unsigned short* __restrict__ vt) { // [b,h,64,s]
  __shared__ unsigned short ldsA[2][128 * KC];
  __shared__ unsigned short ldsB[2][128 * KC];
  int bx = blockIdx.x;
  int which = bx >> 8;
  int rem   = bx & 255;
  int m0 = (rem >> 3) * 128;
  int n0 = (rem & 7) * 128;
  const unsigned short* wt = (which == 0) ? wqt : ((which == 1) ? wkt : wvt);

  v8f c[2][4] = {};
  gemm_128x128(xb, DIM, NB * SEQ, wt, DIM, DIM, m0, n0, ldsA, ldsB, c);

  int wid  = threadIdx.x >> 5;
  int lane = threadIdx.x & 31;
  int waveM = wid >> 1, waveN = wid & 1;
  int half = lane >> 4, col = lane & 15;
#pragma unroll
  for (int mt = 0; mt < 2; ++mt) {
#pragma unroll
    for (int t = 0; t < 4; ++t) {
      int n = n0 + waveN * 64 + t * 16 + col;
      int h = n >> 6, d = n & 63;
#pragma unroll
      for (int v = 0; v < 8; ++v) {
        int m  = m0 + waveM * 32 + mt * 16 + v + half * 8;
        int bb = m >> 11;
        int sq = m & (SEQ - 1);
        size_t bh = (size_t)(bb * NH + h);
        unsigned short val = f2bf(c[mt][t][v]);
        if (which == 0)      qb[(bh * SEQ + sq) * HD + d] = val;
        else if (which == 1) kb[(bh * SEQ + sq) * HD + d] = val;
        else                 vt[(bh * HD + d) * SEQ + sq] = val;
      }
    }
  }
}

// ---------------- banded flash attention (one wave per 16-query tile) ----------------
__global__ __launch_bounds__(256) void k_attn(
    const unsigned short* __restrict__ qb,
    const unsigned short* __restrict__ kb,
    const unsigned short* __restrict__ vt,
    unsigned short* __restrict__ cb) {     // ctx bf16 [b,s,1024]
  __shared__ unsigned short P[8][16 * 32];
  int wid  = threadIdx.x >> 5;
  int lane = threadIdx.x & 31;
  int wave = (blockIdx.x * blockDim.x + threadIdx.x) >> 5;
  int bh = wave >> 7;
  int q0 = (wave & 127) << 4;

  int row  = lane & 15;
  int half = lane >> 4;
  int col  = lane & 15;
  int rowOff = half * 8;
  int ka  = half * 16;
  int kbb = half * 16;

  const unsigned short* qrow = qb + ((size_t)bh * SEQ + (q0 + row)) * HD;
  v8bf ql0 = *(const v8bf*)(qrow + ka);
  v8bf qh0 = *(const v8bf*)(qrow + ka + 8);
  v8bf ql1 = *(const v8bf*)(qrow + 32 + ka);
  v8bf qh1 = *(const v8bf*)(qrow + 32 + ka + 8);
  v16bf aQ0, aQ1;
#pragma unroll
  for (int i = 0; i < 8; ++i) {
    aQ0[i] = ql0[i]; aQ0[8 + i] = qh0[i];
    aQ1[i] = ql1[i]; aQ1[8 + i] = qh1[i];
  }

  v8f acc0 = {}, acc1 = {}, acc2 = {}, acc3 = {};
  float mrow[8], lrow[8];
#pragma unroll
  for (int v = 0; v < 8; ++v) { mrow[v] = NEG_BIG; lrow[v] = 0.f; }

  int js = q0 - WIN; if (js < 0) js = 0; js &= ~31;
  int je = (q0 + 16 + WIN + 31) & ~31; if (je > SEQ) je = SEQ;

  for (int jc = js; jc < je; jc += 32) {
    v8f s0 = {}, s1 = {};
    {
      const unsigned short* kr = kb + ((size_t)bh * SEQ + (jc + col)) * HD;
      v16bf b0 = *(const v16bf*)(kr + kbb);
      v16bf b1 = *(const v16bf*)(kr + 32 + kbb);
      s0 = wmma_bf16(aQ0, b0, s0);
      s0 = wmma_bf16(aQ1, b1, s0);
      kr += (size_t)16 * HD;
      b0 = *(const v16bf*)(kr + kbb);
      b1 = *(const v16bf*)(kr + 32 + kbb);
      s1 = wmma_bf16(aQ0, b0, s1);
      s1 = wmma_bf16(aQ1, b1, s1);
    }
    int j0 = jc + col, j1 = jc + 16 + col;
#pragma unroll
    for (int v = 0; v < 8; ++v) {
      int q = q0 + v + rowOff;
      float a0 = s0[v] * 0.125f;
      float a1 = s1[v] * 0.125f;
      if (j0 < q - WIN || j0 > q + WIN) a0 = NEG_BIG;
      if (j1 < q - WIN || j1 > q + WIN) a1 = NEG_BIG;
      s0[v] = a0; s1[v] = a1;
    }
#pragma unroll
    for (int v = 0; v < 8; ++v) {
      float cmax = fmaxf(s0[v], s1[v]);
#pragma unroll
      for (int off = 8; off >= 1; off >>= 1)
        cmax = fmaxf(cmax, __shfl_xor(cmax, off, 32));
      float mnew = fmaxf(mrow[v], cmax);
      float fac  = __expf(mrow[v] - mnew);
      float e0   = __expf(s0[v] - mnew);
      float e1   = __expf(s1[v] - mnew);
      float esum = e0 + e1;
#pragma unroll
      for (int off = 8; off >= 1; off >>= 1)
        esum += __shfl_xor(esum, off, 32);
      mrow[v] = mnew;
      lrow[v] = lrow[v] * fac + esum;
      acc0[v] *= fac; acc1[v] *= fac; acc2[v] *= fac; acc3[v] *= fac;
      P[wid][(v + rowOff) * 32 + col]      = f2bf(e0);
      P[wid][(v + rowOff) * 32 + 16 + col] = f2bf(e1);
    }
    asm volatile("s_wait_dscnt 0" ::: "memory");
    v8bf pl = *(const v8bf*)&P[wid][row * 32 + ka];
    v8bf ph = *(const v8bf*)&P[wid][row * 32 + ka + 8];
    v16bf aP;
#pragma unroll
    for (int i = 0; i < 8; ++i) { aP[i] = pl[i]; aP[8 + i] = ph[i]; }
    const unsigned short* vr = vt + ((size_t)bh * HD + col) * SEQ + jc + kbb;
    acc0 = wmma_bf16(aP, *(const v16bf*)(vr),            acc0);
    acc1 = wmma_bf16(aP, *(const v16bf*)(vr + 16 * SEQ), acc1);
    acc2 = wmma_bf16(aP, *(const v16bf*)(vr + 32 * SEQ), acc2);
    acc3 = wmma_bf16(aP, *(const v16bf*)(vr + 48 * SEQ), acc3);
  }

  int h = bh & (NH - 1), bb = bh >> 4;
#pragma unroll
  for (int v = 0; v < 8; ++v) {
    float inv = 1.0f / lrow[v];
    int sq = q0 + v + rowOff;
    size_t base = ((size_t)(bb * SEQ + sq)) * DIM + h * HD;
    cb[base + 0 * 16 + col] = f2bf(acc0[v] * inv);
    cb[base + 1 * 16 + col] = f2bf(acc1[v] * inv);
    cb[base + 2 * 16 + col] = f2bf(acc2[v] * inv);
    cb[base + 3 * 16 + col] = f2bf(acc3[v] * inv);
  }
}

// ---------------- output projection + bias ----------------
// grid: 32 (Mblk) x 8 (Nblk) = 256 blocks of 256 threads
__global__ __launch_bounds__(256) void k_outproj(
    const unsigned short* __restrict__ cbf,
    const unsigned short* __restrict__ wot,
    const float* __restrict__ bo,
    float* __restrict__ out) {
  __shared__ unsigned short ldsA[2][128 * KC];
  __shared__ unsigned short ldsB[2][128 * KC];
  int bx = blockIdx.x;
  int m0 = (bx >> 3) * 128;
  int n0 = (bx & 7) * 128;

  v8f c[2][4] = {};
  gemm_128x128(cbf, DIM, NB * SEQ, wot, DIM, DIM, m0, n0, ldsA, ldsB, c);

  int wid  = threadIdx.x >> 5;
  int lane = threadIdx.x & 31;
  int waveM = wid >> 1, waveN = wid & 1;
  int half = lane >> 4, col = lane & 15;
#pragma unroll
  for (int mt = 0; mt < 2; ++mt) {
#pragma unroll
    for (int t = 0; t < 4; ++t) {
      int n = n0 + waveN * 64 + t * 16 + col;
      float bias = bo[n];
#pragma unroll
      for (int v = 0; v < 8; ++v) {
        int m = m0 + waveM * 32 + mt * 16 + v + half * 8;
        out[(size_t)m * DIM + n] = c[mt][t][v] + bias;
      }
    }
  }
}

// ---------------- host side ----------------
extern "C" void kernel_launch(void* const* d_in, const int* in_sizes, int n_in,
                              void* d_out, int out_size, void* d_ws, size_t ws_size,
                              hipStream_t stream) {
  const float* x  = (const float*)d_in[0];
  const float* Wq = (const float*)d_in[1];
  const float* Wk = (const float*)d_in[2];
  const float* Wv = (const float*)d_in[3];
  const float* Wo = (const float*)d_in[4];
  const float* bo = (const float*)d_in[5];
  float* out = (float*)d_out;
  (void)in_sizes; (void)n_in; (void)out_size; (void)ws_size;

  const size_t SZ_X = (size_t)NB * SEQ * DIM;
  const size_t SZ_W = (size_t)DIM * DIM;
  unsigned short* Xb  = (unsigned short*)d_ws;
  unsigned short* WqT = Xb  + SZ_X;
  unsigned short* WkT = WqT + SZ_W;
  unsigned short* WvT = WkT + SZ_W;
  unsigned short* WoT = WvT + SZ_W;
  unsigned short* Qb  = WoT + SZ_W;
  unsigned short* Kb  = Qb  + SZ_X;
  unsigned short* Vt  = Kb  + SZ_X;
  unsigned short* Cb  = Vt  + SZ_X;

  k_cvt_x<<<(int)(SZ_X / 256), 256, 0, stream>>>(x, Xb);
  dim3 tg(32, 32);
  k_cvt_wT<<<tg, 256, 0, stream>>>(Wq, WqT);
  k_cvt_wT<<<tg, 256, 0, stream>>>(Wk, WkT);
  k_cvt_wT<<<tg, 256, 0, stream>>>(Wv, WvT);
  k_cvt_wT<<<tg, 256, 0, stream>>>(Wo, WoT);

  k_proj<<<768, 256, 0, stream>>>(Xb, WqT, WkT, WvT, Qb, Kb, Vt);
  k_attn<<<512, 256, 0, stream>>>(Qb, Kb, Vt, Cb);
  k_outproj<<<256, 256, 0, stream>>>(Cb, WoT, bo, out);
}
